// CausalSelfAttention_21895743275608
// MI455X (gfx1250) — matrix operand order repaired and checked by reading, not compiled, not run on hardware
//
#include <hip/hip_runtime.h>

// ---------------------------------------------------------------------------
// Causal self-attention for MI455X (gfx1250): f16 WMMA pipeline.
//   K0: fp32 -> f16 conversions (x) and transposed conversions (W_qkv, W_proj)
//   K1: QKV GEMM (WMMA 16x16x32 f16, double-buffered async-LDS staging)
//   K2: flash attention (WMMA S=QK^T, online softmax, WMMA O+=PV)
//   K3: projection GEMM (WMMA), bias, fp32 out
// ---------------------------------------------------------------------------

#define DEV __device__ __forceinline__

typedef __attribute__((ext_vector_type(4)))  _Float16 v4h;
typedef __attribute__((ext_vector_type(8)))  _Float16 v8h;
typedef __attribute__((ext_vector_type(16))) _Float16 v16h;
typedef __attribute__((ext_vector_type(8)))  float    v8f;
typedef int v4i __attribute__((vector_size(16)));

union Frag { v16h v; v8h h[2]; };

constexpr int BB = 4;        // batch
constexpr int TT = 2048;     // sequence
constexpr int CC = 1024;     // embed
constexpr int HH = 16;       // heads
constexpr int HS = 64;       // head size
constexpr int M1 = BB * TT;  // 8192 rows
constexpr int N1 = 3 * CC;   // 3072
constexpr int K1 = CC;       // 1024

DEV v8h ld8(const _Float16* p) { return *reinterpret_cast<const v8h*>(p); }
DEV void st8(_Float16* p, v8h x) { *reinterpret_cast<v8h*>(p) = x; }

// ---- async global -> LDS copy (CDNA5 GLOBAL_LOAD_ASYNC_TO_LDS_B128) -------
#define AS1 __attribute__((address_space(1)))
#define AS3 __attribute__((address_space(3)))

#if defined(__AMDGCN__) && __has_builtin(__builtin_amdgcn_global_load_async_to_lds_b128)
#define HAVE_ASYNC 1
#else
#define HAVE_ASYNC 0
#endif

// copy 16 bytes (8 halfs) global -> LDS
DEV void cp16(_Float16* ldst, const _Float16* gsrc) {
#if HAVE_ASYNC
    __builtin_amdgcn_global_load_async_to_lds_b128(
        (AS1 v4i*)gsrc, (AS3 v4i*)ldst, /*offset=*/0, /*cpol=*/0);
#else
    st8(ldst, ld8(gsrc));
#endif
}

DEV void wait_async() {
#if HAVE_ASYNC
#if __has_builtin(__builtin_amdgcn_s_wait_asynccnt)
    __builtin_amdgcn_s_wait_asynccnt(0);
#else
    asm volatile("s_wait_asynccnt 0x0" ::: "memory");
#endif
#endif
}

DEV v8f wmma_f16(v16h a, v16h b, v8f c) {
    // D = A(16x32 f16) * B(32x16 f16) + C(16x16 f32)
    return __builtin_amdgcn_wmma_f32_16x16x32_f16(
        /*neg_a=*/false, a, /*neg_b=*/false, b,
        /*c_mod=*/(short)0, c, /*reuse_a=*/false, /*reuse_b=*/false);
}

// ---------------------------------------------------------------------------
// K0: conversions
// ---------------------------------------------------------------------------
__global__ void cvt_f32_f16(const float* __restrict__ in,
                            _Float16* __restrict__ out, long n) {
    long i = (long)blockIdx.x * blockDim.x + threadIdx.x;
    long stride = (long)gridDim.x * blockDim.x;
    for (; i < n; i += stride) out[i] = (_Float16)in[i];
}

// in: [K][N] row-major fp32  ->  out: [N][K] f16 (transposed)
__global__ void cvt_tr_f16(const float* __restrict__ in,
                           _Float16* __restrict__ out, int K, int N) {
    long total = (long)K * N;
    long i = (long)blockIdx.x * blockDim.x + threadIdx.x;
    long stride = (long)gridDim.x * blockDim.x;
    for (; i < total; i += stride) {
        int k = (int)(i / N), n = (int)(i % N);
        out[(long)n * K + k] = (_Float16)in[i];
    }
}

// ---------------------------------------------------------------------------
// K1 / K3: tiled WMMA GEMM.  A:[M1][K1] f16, Bt:[N][K1] f16 (pre-transposed).
// Block tile 128x128, BK=32, 8 waves; each wave: 4(M) x 2(N) wmma tiles.
// Double-buffered LDS; next tile staged via async-LDS DMA while WMMAs run.
// EPI==0: out[m*N+n] = acc + bias[n]  (fp32)
// EPI==1: qkv scatter to [B,H,T,hs] f16, q scaled by 1/sqrt(hs)
// ---------------------------------------------------------------------------
template <int N, int EPI>
__global__ __launch_bounds__(256)
void gemm_wmma(const _Float16* __restrict__ A,
               const _Float16* __restrict__ Bt,
               const float* __restrict__ bias,
               float* __restrict__ outF,
               _Float16* __restrict__ qb,
               _Float16* __restrict__ kb,
               _Float16* __restrict__ vb) {
    constexpr int LDT = 40;  // padded row stride (halfs), 80B = 5*16B
    __shared__ _Float16 lA[2][128 * LDT];
    __shared__ _Float16 lB[2][128 * LDT];

    const int tid  = threadIdx.x;
    const int wave = tid >> 5, lane = tid & 31;
    const int hl = lane >> 4, ln = lane & 15;
    const int mBase = blockIdx.y * 128;
    const int nBase = blockIdx.x * 128;
    const int wm = (wave >> 2) * 64;  // wave M offset in block
    const int wn = (wave & 3) * 32;   // wave N offset in block

    // staging assignment: 256 threads cover 128 rows x 32 halfs per tile
    const int srow = tid >> 1;
    const int sc16 = (tid & 1) * 16;
    const _Float16* gA = A + (long)(mBase + srow) * K1 + sc16;
    const _Float16* gB = Bt + (long)(nBase + srow) * K1 + sc16;

    v8f acc[4][2];
    for (int i = 0; i < 4; ++i)
        for (int j = 0; j < 2; ++j) acc[i][j] = (v8f)0.0f;

    auto stage = [&](int buf, int kb0) {
        cp16(&lA[buf][srow * LDT + sc16], gA + kb0);
        cp16(&lA[buf][srow * LDT + sc16 + 8], gA + kb0 + 8);
        cp16(&lB[buf][srow * LDT + sc16], gB + kb0);
        cp16(&lB[buf][srow * LDT + sc16 + 8], gB + kb0 + 8);
    };

    stage(0, 0);
    wait_async();
    __syncthreads();

    int cur = 0;
    for (int kb0 = 0; kb0 < K1; kb0 += 32) {
        if (kb0 + 32 < K1) stage(cur ^ 1, kb0 + 32);  // prefetch next tile

        Frag af[4], bf[2];
        for (int mt = 0; mt < 4; ++mt) {
            // A 16x32 layout: chunk0 @ k=(L/16)*8, chunk1 @ k=16+(L/16)*8
            const _Float16* p = &lA[cur][(wm + mt * 16 + ln) * LDT + hl * 8];
            af[mt].h[0] = ld8(p);
            af[mt].h[1] = ld8(p + 16);
        }
        for (int nt = 0; nt < 2; ++nt) {
            // B 32x16 layout: chunk0 @ k=(L/16)*8, chunk1 @ k=16+(L/16)*8, column n=L%16
            const _Float16* p = &lB[cur][(wn + nt * 16 + ln) * LDT + hl * 8];
            bf[nt].h[0] = ld8(p);
            bf[nt].h[1] = ld8(p + 16);
        }
        for (int mt = 0; mt < 4; ++mt)
            for (int nt = 0; nt < 2; ++nt)
                acc[mt][nt] = wmma_f16(af[mt].v, bf[nt].v, acc[mt][nt]);

        wait_async();
        __syncthreads();
        cur ^= 1;
    }

    // epilogue: C/D layout -> lane: n = L%16, m = r + 8*(L/16)
    for (int nt = 0; nt < 2; ++nt) {
        const int n = nBase + wn + nt * 16 + ln;
        const float bs = bias[n];
        for (int mt = 0; mt < 4; ++mt) {
            for (int r = 0; r < 8; ++r) {
                const int m = mBase + wm + mt * 16 + r + hl * 8;
                const float val = acc[mt][nt][r] + bs;
                if (EPI == 0) {
                    outF[(long)m * N + n] = val;
                } else {
                    const int part = n >> 10;        // 0=q 1=k 2=v
                    const int c = n & 1023;
                    const int h = c >> 6, d = c & 63;
                    const int b = m >> 11, t = m & 2047;
                    const long dst = (((long)(b * HH + h) * TT) + t) * HS + d;
                    if (part == 0)      qb[dst] = (_Float16)(val * 0.125f);
                    else if (part == 1) kb[dst] = (_Float16)val;
                    else                vb[dst] = (_Float16)val;
                }
            }
        }
    }
}

// ---------------------------------------------------------------------------
// K2: flash attention. Grid (B*H, T/128). 8 waves; wave w owns 16 q rows.
// q/k/v layout: [B*H][T][HS] f16; q pre-scaled by 1/sqrt(HS).
// K tile staged via async-LDS DMA (overlaps the V transpose); V transposed
// through VGPRs; P converted C-layout -> A-layout via per-wave LDS scratch.
// ---------------------------------------------------------------------------
__global__ __launch_bounds__(256)
void flash_attn(const _Float16* __restrict__ qg,
                const _Float16* __restrict__ kg,
                const _Float16* __restrict__ vg,
                _Float16* __restrict__ aout) {
    extern __shared__ _Float16 smem[];
    constexpr int LDK = 72;   // K tile row stride (halfs), 144B
    constexpr int LDV = 136;  // Vt row stride, 272B
    constexpr int LDP = 136;  // per-wave P row stride
    _Float16* Kl = smem;                 // [128][LDK]
    _Float16* Vt = Kl + 128 * LDK;       // [64][LDV]   (V transposed: [d][kj])
    _Float16* Pl = Vt + 64 * LDV;        // [8][16][LDP] per-wave scratch

    const int tid  = threadIdx.x;
    const int wave = tid >> 5, lane = tid & 31;
    const int hl = lane >> 4, ln = lane & 15;
    const int bh = blockIdx.x;           // b*H + h
    const int qt = blockIdx.y;           // query tile
    const int q0w = qt * 128 + wave * 16;

    const _Float16* qp = qg + (long)bh * TT * HS;
    const _Float16* kp = kg + (long)bh * TT * HS;
    const _Float16* vp = vg + (long)bh * TT * HS;

    // Q fragments held in registers for the whole kernel (2 k-chunks of 32)
    Frag aq[2];
    {
        const _Float16* qrow = qp + (long)(q0w + ln) * HS;
        for (int kc = 0; kc < 2; ++kc) {
            aq[kc].h[0] = ld8(qrow + kc * 32 + hl * 8);
            aq[kc].h[1] = ld8(qrow + kc * 32 + 16 + hl * 8);
        }
    }

    float m_i[8], l_i[8];
    v8f o[4];
    for (int r = 0; r < 8; ++r) { m_i[r] = -1e30f; l_i[r] = 0.0f; }
    for (int dt = 0; dt < 4; ++dt) o[dt] = (v8f)0.0f;

    _Float16* Pw = Pl + wave * 16 * LDP;

    for (int kvt = 0; kvt <= qt; ++kvt) {
        const int kb0 = kvt * 128;
        // ---- stage K tile via async DMA (overlaps V transpose below) ----
        for (int i = 0; i < 2; ++i) {
            int idx = tid + i * 256;         // 512 chunks of 16 halfs
            int row = idx >> 2;
            int c16 = (idx & 3) * 16;
            const _Float16* g = kp + (long)(kb0 + row) * HS + c16;
            cp16(&Kl[row * LDK + c16], g);
            cp16(&Kl[row * LDK + c16 + 8], g + 8);
        }
        // ---- V transposed into LDS through VGPRs ----
        for (int i = 0; i < 8; ++i) {
            int idx = tid + i * 256;         // 2048 quads
            int kj = idx >> 4;
            int d0 = (idx & 15) * 4;
            v4h x = *reinterpret_cast<const v4h*>(vp + (long)(kb0 + kj) * HS + d0);
            Vt[(d0 + 0) * LDV + kj] = x[0];
            Vt[(d0 + 1) * LDV + kj] = x[1];
            Vt[(d0 + 2) * LDV + kj] = x[2];
            Vt[(d0 + 3) * LDV + kj] = x[3];
        }
        wait_async();
        __syncthreads();

        // ---- S = Q K^T  (16 x 128, q already scaled) ----
        v8f s[8];
        for (int nt = 0; nt < 8; ++nt) {
            s[nt] = (v8f)0.0f;
            for (int kc = 0; kc < 2; ++kc) {
                Frag bk;
                const _Float16* p = &Kl[(nt * 16 + ln) * LDK + kc * 32 + hl * 8];
                bk.h[0] = ld8(p);
                bk.h[1] = ld8(p + 16);
                s[nt] = wmma_f16(aq[kc].v, bk.v, s[nt]);
            }
        }
        // causal mask, only the diagonal tile
        if (kvt == qt) {
            for (int nt = 0; nt < 8; ++nt) {
                const int kgl = kb0 + nt * 16 + ln;
                for (int r = 0; r < 8; ++r) {
                    const int qgl = q0w + r + hl * 8;
                    if (kgl > qgl) s[nt][r] = -1e30f;
                }
            }
        }

        // ---- online softmax (row reductions across 16-lane half-groups) ----
        float mnew[8], rs[8];
        for (int r = 0; r < 8; ++r) {
            float pm = s[0][r];
            for (int nt = 1; nt < 8; ++nt) pm = fmaxf(pm, s[nt][r]);
            pm = fmaxf(pm, __shfl_xor(pm, 1));
            pm = fmaxf(pm, __shfl_xor(pm, 2));
            pm = fmaxf(pm, __shfl_xor(pm, 4));
            pm = fmaxf(pm, __shfl_xor(pm, 8));
            mnew[r] = fmaxf(m_i[r], pm);
        }
        for (int r = 0; r < 8; ++r) {
            const float sc = __expf(m_i[r] - mnew[r]);
            l_i[r] *= sc;
            for (int dt = 0; dt < 4; ++dt) o[dt][r] *= sc;
            m_i[r] = mnew[r];
            rs[r] = 0.0f;
        }
        for (int nt = 0; nt < 8; ++nt) {
            for (int r = 0; r < 8; ++r) {
                const float p = __expf(s[nt][r] - mnew[r]);
                rs[r] += p;
                // C-layout -> A-layout via per-wave LDS scratch
                Pw[(r + hl * 8) * LDP + nt * 16 + ln] = (_Float16)p;
            }
        }
        for (int r = 0; r < 8; ++r) {
            float sum = rs[r];
            sum += __shfl_xor(sum, 1);
            sum += __shfl_xor(sum, 2);
            sum += __shfl_xor(sum, 4);
            sum += __shfl_xor(sum, 8);
            l_i[r] += sum;
        }

        // ---- O += P V  (P: 16x128 f16, Vt: [d][kj]) ----
        for (int kc = 0; kc < 4; ++kc) {
            Frag ap;
            const _Float16* pp = &Pw[ln * LDP + kc * 32 + hl * 8];
            ap.h[0] = ld8(pp);
            ap.h[1] = ld8(pp + 16);
            for (int dt = 0; dt < 4; ++dt) {
                Frag bv;
                const _Float16* pv = &Vt[(dt * 16 + ln) * LDV + kc * 32 + hl * 8];
                bv.h[0] = ld8(pv);
                bv.h[1] = ld8(pv + 16);
                o[dt] = wmma_f16(ap.v, bv.v, o[dt]);
            }
        }
        __syncthreads();
    }

    // ---- epilogue: normalize, write [B,T,C] f16 for projection GEMM ----
    const int b = bh >> 4, h = bh & 15;
    for (int dt = 0; dt < 4; ++dt) {
        for (int r = 0; r < 8; ++r) {
            const int qgl = q0w + r + hl * 8;
            const int d = dt * 16 + ln;
            const float val = o[dt][r] / l_i[r];
            aout[((long)(b * TT + qgl)) * CC + h * HS + d] = (_Float16)val;
        }
    }
}

// ---------------------------------------------------------------------------
// Host-side orchestration
// ---------------------------------------------------------------------------
extern "C" void kernel_launch(void* const* d_in, const int* in_sizes, int n_in,
                              void* d_out, int out_size, void* d_ws, size_t ws_size,
                              hipStream_t stream) {
    (void)in_sizes; (void)n_in; (void)out_size; (void)ws_size;
    const float* x     = (const float*)d_in[0];
    const float* Wqkv  = (const float*)d_in[1];
    const float* bqkv  = (const float*)d_in[2];
    const float* Wproj = (const float*)d_in[3];
    const float* bproj = (const float*)d_in[4];
    float* out = (float*)d_out;

    char* ws = (char*)d_ws;
    size_t off = 0;
    auto carve = [&](size_t bytes) -> void* {
        void* p = ws + off;
        off += (bytes + 255) & ~(size_t)255;
        return p;
    };
    _Float16* xh   = (_Float16*)carve((size_t)M1 * K1 * 2);  // x in f16
    _Float16* wq_t = (_Float16*)carve((size_t)N1 * K1 * 2);  // W_qkv^T f16
    _Float16* wp_t = (_Float16*)carve((size_t)CC * CC * 2);  // W_proj^T f16
    _Float16* qb   = (_Float16*)carve((size_t)M1 * CC * 2);  // [B,H,T,hs]
    _Float16* kb   = (_Float16*)carve((size_t)M1 * CC * 2);
    _Float16* vb   = (_Float16*)carve((size_t)M1 * CC * 2);
    _Float16* aout = (_Float16*)carve((size_t)M1 * CC * 2);  // attn out f16

    cvt_f32_f16<<<2048, 256, 0, stream>>>(x, xh, (long)M1 * K1);
    cvt_tr_f16<<<2048, 256, 0, stream>>>(Wqkv, wq_t, K1, N1);
    cvt_tr_f16<<<2048, 256, 0, stream>>>(Wproj, wp_t, CC, CC);

    dim3 g1(N1 / 128, M1 / 128);
    gemm_wmma<N1, 1><<<g1, 256, 0, stream>>>(xh, wq_t, bqkv, nullptr, qb, kb, vb);

    dim3 gf(BB * HH, TT / 128);
    const size_t shbytes = (size_t)(128 * 72 + 64 * 136 + 8 * 16 * 136) * sizeof(_Float16);
    flash_attn<<<gf, 256, shbytes, stream>>>(qb, kb, vb, aout);

    dim3 g2(CC / 128, M1 / 128);
    gemm_wmma<CC, 0><<<g2, 256, 0, stream>>>(aout, wp_t, bproj, out, nullptr, nullptr, nullptr);
}
